// AttentionPooling_43499428774429
// MI455X (gfx1250) — matrix operand order repaired and checked by reading, not compiled, not run on hardware
//
#include <hip/hip_runtime.h>

#define H   256
#define S   32
#define NH  8
#define HD  32
#define NB  16

typedef __attribute__((ext_vector_type(16))) __bf16          v16bf;
typedef __attribute__((ext_vector_type(8)))  float           v8f;
typedef __attribute__((ext_vector_type(16))) unsigned short  u16x16;
typedef __attribute__((ext_vector_type(4)))  unsigned int    u32x4;

union FragU {
  u16x16 v;
  unsigned short s[16];
  u32x4 q[2];
};

static __device__ __forceinline__ unsigned short f2bf(float f) {
  unsigned u = __builtin_bit_cast(unsigned, f);
  u += 0x7FFFu + ((u >> 16) & 1u);           // round-to-nearest-even
  return (unsigned short)(u >> 16);
}

static __device__ __forceinline__ v8f wmma_bf16(const FragU& a, const FragU& b, v8f c) {
  return __builtin_amdgcn_wmma_f32_16x16x32_bf16(
      false, __builtin_bit_cast(v16bf, a.v),
      false, __builtin_bit_cast(v16bf, b.v),
      (short)0, c, false, false);
}

// ---------------------------------------------------------------------------
// Kernel 1: q = seed @ Wq (in LDS), then build combined B matrix, K-transposed:
//   BmatT[c][k]      = bf16( (1/sqrt(hd)) * sum_d Wk[k][h*32+d] * q[s][h*32+d] ),  c = s*8+h
//   BmatT[256+j][k]  = bf16( Wv[k][j] )
// ---------------------------------------------------------------------------
__global__ __launch_bounds__(256)
void prep_kernel(const float* __restrict__ seed, const float* __restrict__ Wq,
                 const float* __restrict__ Wk, const float* __restrict__ Wv,
                 unsigned short* __restrict__ BmatT) {
  __shared__ float qs[S * H];
  const int t = threadIdx.x;
  for (int s = 0; s < S; ++s) {
    float acc = 0.f;
    for (int k = 0; k < H; ++k) acc += seed[s * H + k] * Wq[k * H + t];
    qs[s * H + t] = acc;
  }
  __syncthreads();
  const float scale = 0.17677669529663687f;  // 1/sqrt(32)
  for (int c = 0; c < H; ++c) {
    const int s = c >> 3, h = c & 7;
    const float* wkrow = Wk + t * H + h * HD;
    const float* qrow  = qs + s * H + h * HD;
    float acc = 0.f;
    for (int d = 0; d < HD; ++d) acc += wkrow[d] * qrow[d];
    BmatT[c * H + t] = f2bf(acc * scale);
  }
  for (int j = 0; j < H; ++j) BmatT[(H + j) * H + t] = f2bf(Wv[t * H + j]);
}

// ---------------------------------------------------------------------------
// Kernel 2: edge_features f32 -> bf16 (row-major [E][256])
// ---------------------------------------------------------------------------
__global__ __launch_bounds__(256)
void cvt_kernel(const float* __restrict__ in, unsigned short* __restrict__ out, int n4) {
  int i = blockIdx.x * 256 + threadIdx.x;
  if (i >= n4) return;
  float4 v = ((const float4*)in)[i];
  ushort4 o;
  o.x = f2bf(v.x); o.y = f2bf(v.y); o.z = f2bf(v.z); o.w = f2bf(v.w);
  ((ushort4*)out)[i] = o;
}

// ---------------------------------------------------------------------------
// Kernel 3: fused scores+V GEMM:  C[E][512] = Afeat[E][256] @ B[256][512]
//   cols [0,256)   -> scores f32   (score scale folded into P)
//   cols [256,512) -> V in bf16 (for the attention WMMA stage)
// Block = 256 thr (8 waves, 4Mx2N), block tile 128x128, wave tile 32x64.
// ---------------------------------------------------------------------------
__global__ __launch_bounds__(256)
void gemm_kernel(const unsigned short* __restrict__ A, const unsigned short* __restrict__ Bm,
                 float* __restrict__ scores, unsigned short* __restrict__ Vbf) {
  const int wave = threadIdx.x >> 5, lane = threadIdx.x & 31;
  const int lm = lane & 15, lh = lane >> 4;
  const int Mb = blockIdx.x * 128 + (wave & 3) * 32;
  const int Nb = blockIdx.y * 128 + (wave >> 2) * 64;

  const v8f zero = {0.f, 0.f, 0.f, 0.f, 0.f, 0.f, 0.f, 0.f};
  v8f acc[2][4];
  for (int m = 0; m < 2; ++m)
    for (int n = 0; n < 4; ++n) acc[m][n] = zero;

  for (int kk = 0; kk < 8; ++kk) {   // K = 256, 32 per WMMA
    FragU a[2], b[4];
    for (int m = 0; m < 2; ++m) {
      // A 16x32 bf16: lane lm holds row M, K = (e/8)*16 + lh*8 + e%8
      const unsigned short* p = A + (size_t)(Mb + m * 16 + lm) * H + kk * 32 + lh * 8;
      a[m].q[0] = *(const u32x4*)p;
      a[m].q[1] = *(const u32x4*)(p + 16);
    }
    for (int n = 0; n < 4; ++n) {
      // B 32x16 bf16 (K-transposed storage): lane lm = col N, K = (e/8)*16 + lh*8 + e%8
      const unsigned short* p = Bm + (size_t)(Nb + n * 16 + lm) * H + kk * 32 + lh * 8;
      b[n].q[0] = *(const u32x4*)p;
      b[n].q[1] = *(const u32x4*)(p + 16);
    }
    for (int m = 0; m < 2; ++m)
      for (int n = 0; n < 4; ++n) acc[m][n] = wmma_bf16(a[m], b[n], acc[m][n]);
  }

  for (int m = 0; m < 2; ++m)
    for (int n = 0; n < 4; ++n) {
      const int col   = Nb + n * 16 + lm;
      const int rbase = Mb + m * 16 + lh * 8;   // VGPR r -> row rbase+r
      if (col < H) {
        for (int r = 0; r < 8; ++r) scores[(size_t)(rbase + r) * H + col] = acc[m][n][r];
      } else {
        const int vc = col - H;
        for (int r = 0; r < 8; ++r) Vbf[(size_t)(rbase + r) * H + vc] = f2bf(acc[m][n][r]);
      }
    }
}

// ---------------------------------------------------------------------------
// Kernel 4: per-graph online segment softmax stats (thread = column, coalesced)
// ---------------------------------------------------------------------------
__global__ __launch_bounds__(256)
void softmax_kernel(const float* __restrict__ scores, float* __restrict__ mbuf,
                    float* __restrict__ rden, int epg) {
  const int b = blockIdx.x, c = threadIdx.x;
  const float* p = scores + (size_t)b * epg * H + c;
  float mx = -3.402823466e38f, s = 0.f;
  for (int e = 0; e < epg; ++e) {
    float x = p[(size_t)e * H];
    float nm = fmaxf(mx, x);
    s = s * __expf(mx - nm) + __expf(x - nm);
    mx = nm;
  }
  mbuf[b * H + c] = mx;
  rden[b * H + c] = 1.f / s;
}

// ---------------------------------------------------------------------------
// Kernel 5: attention accumulate. One wave per (graph, head):
//   att[b, s, h*32+d] = sum_e w[e, s, h] * V[e, h*32+d]
//   A = w^T (32 x 1024, exp on the fly), B = V_h (1024 x 32), 32 K-steps of WMMA.
// ---------------------------------------------------------------------------
__global__ __launch_bounds__(32)
void att_kernel(const float* __restrict__ scores, const unsigned short* __restrict__ Vbf,
                const float* __restrict__ mbuf, const float* __restrict__ rden,
                float* __restrict__ att, int epg) {
  const int b = blockIdx.x >> 3, h = blockIdx.x & 7;
  const int lane = threadIdx.x, lm = lane & 15, lh = lane >> 4;
  const int c0 = lm * NH + h, c1 = (lm + 16) * NH + h;
  const float m0 = mbuf[b * H + c0], r0 = rden[b * H + c0];
  const float m1 = mbuf[b * H + c1], r1 = rden[b * H + c1];
  const float* sb = scores + (size_t)b * epg * H;
  const unsigned short* vb = Vbf + (size_t)b * epg * H + h * HD;

  const v8f zero = {0.f, 0.f, 0.f, 0.f, 0.f, 0.f, 0.f, 0.f};
  v8f acc[2][2] = {{zero, zero}, {zero, zero}};

  const int ksteps = epg >> 5;
  for (int kk = 0; kk < ksteps; ++kk) {
    const int eb = kk * 32;
    FragU a0, a1, b0, b1;
    for (int ei = 0; ei < 16; ++ei) {
      int e = eb + ((ei >> 3) << 4) + lh * 8 + (ei & 7);   // A layout K index
      a0.s[ei] = f2bf(__expf(sb[(size_t)e * H + c0] - m0) * r0);
      a1.s[ei] = f2bf(__expf(sb[(size_t)e * H + c1] - m1) * r1);
    }
    for (int ei = 0; ei < 16; ++ei) {
      int e = eb + ((ei >> 3) << 4) + lh * 8 + (ei & 7);   // B layout K index
      b0.s[ei] = vb[(size_t)e * H + lm];
      b1.s[ei] = vb[(size_t)e * H + 16 + lm];
    }
    acc[0][0] = wmma_bf16(a0, b0, acc[0][0]);
    acc[0][1] = wmma_bf16(a0, b1, acc[0][1]);
    acc[1][0] = wmma_bf16(a1, b0, acc[1][0]);
    acc[1][1] = wmma_bf16(a1, b1, acc[1][1]);
  }

  float* ob = att + (size_t)b * S * H + h * HD;
  for (int m = 0; m < 2; ++m)
    for (int n = 0; n < 2; ++n)
      for (int r = 0; r < 8; ++r) {
        int s = m * 16 + lh * 8 + r;
        ob[s * H + n * 16 + lm] = acc[m][n][r];
      }
}

// ---------------------------------------------------------------------------
// Kernel 6: per graph: y = seed + att@Wo + bo; LayerNorm; silu(flat@W1+b1)@W2+b2
// ---------------------------------------------------------------------------
__global__ __launch_bounds__(256)
void out_kernel(const float* __restrict__ att, const float* __restrict__ seed,
                const float* __restrict__ Wo, const float* __restrict__ bo,
                const float* __restrict__ ln_g, const float* __restrict__ ln_b,
                const float* __restrict__ W1, const float* __restrict__ b1,
                const float* __restrict__ W2, const float* __restrict__ b2,
                float* __restrict__ out) {
  __shared__ float yflat[S * H];
  __shared__ float red[H];
  __shared__ float rowbuf[H];
  const int b = blockIdx.x, j = threadIdx.x;

  for (int s = 0; s < S; ++s) {
    rowbuf[j] = att[((size_t)b * S + s) * H + j];
    __syncthreads();
    float acc = seed[s * H + j] + bo[j];
    for (int k = 0; k < H; ++k) acc += rowbuf[k] * Wo[k * H + j];
    red[j] = acc;
    __syncthreads();
    for (int o = 128; o > 0; o >>= 1) { if (j < o) red[j] += red[j + o]; __syncthreads(); }
    const float mu = red[0] * (1.f / H);
    __syncthreads();
    const float d = acc - mu;
    red[j] = d * d;
    __syncthreads();
    for (int o = 128; o > 0; o >>= 1) { if (j < o) red[j] += red[j + o]; __syncthreads(); }
    const float var = red[0] * (1.f / H);
    __syncthreads();
    yflat[s * H + j] = d * rsqrtf(var + 1e-5f) * ln_g[j] + ln_b[j];
    __syncthreads();
  }

  float acc = b1[j];
  for (int i = 0; i < S * H; ++i) acc += yflat[i] * W1[i * H + j];
  const float h1 = acc / (1.f + __expf(-acc));   // silu
  red[j] = h1;
  __syncthreads();
  float o = b2[j];
  for (int k = 0; k < H; ++k) o += red[k] * W2[k * H + j];
  out[b * H + j] = o;
}

// ---------------------------------------------------------------------------
extern "C" void kernel_launch(void* const* d_in, const int* in_sizes, int n_in,
                              void* d_out, int out_size, void* d_ws, size_t ws_size,
                              hipStream_t stream) {
  const float* edge_features = (const float*)d_in[0];
  // d_in[1] edge_coords: unused by reference. d_in[2] batch: equal sorted segments.
  const float* seed = (const float*)d_in[3];
  const float* Wq   = (const float*)d_in[4];
  const float* Wk   = (const float*)d_in[5];
  const float* Wv   = (const float*)d_in[6];
  const float* Wo   = (const float*)d_in[7];
  const float* bo   = (const float*)d_in[8];
  const float* ln_g = (const float*)d_in[9];
  const float* ln_b = (const float*)d_in[10];
  const float* W1   = (const float*)d_in[11];
  const float* b1   = (const float*)d_in[12];
  const float* W2   = (const float*)d_in[13];
  const float* b2   = (const float*)d_in[14];
  float* out = (float*)d_out;

  const int E   = in_sizes[0] / H;     // 16384
  const int epg = E / NB;              // 1024 edges per graph

  char* w = (char*)d_ws;
  unsigned short* BmatT  = (unsigned short*)w;                                  // 256 KB
  unsigned short* Afeat  = (unsigned short*)(w + (256u << 10));                 // 8 MB
  float*          scores = (float*)(w + (256u << 10) + (8u << 20));             // 16 MB
  unsigned short* Vbf    = (unsigned short*)(w + (256u << 10) + (24u << 20));   // 8 MB
  float*          mbuf   = (float*)(w + (256u << 10) + (32u << 20));
  float*          rden   = mbuf + NB * H;
  float*          att    = rden + NB * H;                                       // 512 KB

  prep_kernel<<<1, 256, 0, stream>>>(seed, Wq, Wk, Wv, BmatT);

  const int n4 = E * H / 4;
  cvt_kernel<<<(n4 + 255) / 256, 256, 0, stream>>>(edge_features, Afeat, n4);

  dim3 g(E / 128, 4);
  gemm_kernel<<<g, 256, 0, stream>>>(Afeat, BmatT, scores, Vbf);

  softmax_kernel<<<NB, 256, 0, stream>>>(scores, mbuf, rden, epg);

  att_kernel<<<NB * NH, 32, 0, stream>>>(scores, Vbf, mbuf, rden, att, epg);

  out_kernel<<<NB, 256, 0, stream>>>(att, seed, Wo, bo, ln_g, ln_b, W1, b1, W2, b2, out);
}
